// XEyTransformerLayer_75539884802877
// MI455X (gfx1250) — matrix operand order repaired and checked by reading, not compiled, not run on hardware
//
#include <hip/hip_runtime.h>
#include <hip/hip_bf16.h>

typedef __attribute__((ext_vector_type(8)))  float  v8f;
typedef __attribute__((ext_vector_type(16))) __bf16 v16bf;

#define NNODE 256
#define EPSV 1e-5f

// param leaf indices in d_in: 6 + 2*sorted_key_pos (W, then b)
enum {
  P_E_ADD=6, P_E_MUL=8, P_E_OUT=10, P_E_REG_ADD=12, P_E_REG_MUL=14, P_E_Y=16,
  P_K=18, P_LINE1=20, P_LINE2=22, P_LINX1=24, P_LINX2=26, P_LINY1=28, P_LINY2=30,
  P_NORME1=32, P_NORME2=34, P_NORMX1=36, P_NORMX2=38, P_NORMY1=40, P_NORMY2=42,
  P_Q=44, P_V=46, P_X_OUT=48, P_X_Y=50, P_Y_E_ADD=52, P_Y_E_MUL=54,
  P_Y_OUT1=56, P_Y_OUT2=58, P_Y_X_ADD=60, P_Y_X_MUL=62, P_Y_Y=64
};

// ---------------- helpers ----------------
__device__ __forceinline__ unsigned bfbits(float f) {
  unsigned u = __builtin_bit_cast(unsigned, f);
  return (u + 0x7FFFu + ((u >> 16) & 1u)) >> 16;
}
__device__ __forceinline__ unsigned pack2(float lo, float hi) {
  return bfbits(lo) | (bfbits(hi) << 16);
}
__device__ __forceinline__ v8f z8() {
  v8f z;
#pragma unroll
  for (int i = 0; i < 8; i++) z[i] = 0.f;
  return z;
}
__device__ __forceinline__ v8f wmma_bf(v16bf a, v16bf b, v8f c) {
  return __builtin_amdgcn_wmma_f32_16x16x32_bf16(false, a, false, b, (short)0, c, false, false);
}
union FragU { v16bf v; uint4 q[2]; unsigned u[8]; };

// A operand (16x32 bf16) from LDS row-major [m][K] tile. p -> &row[(kb+16*half)*2]
__device__ __forceinline__ v16bf afrag_s(const char* p) {
  FragU f;
  f.q[0] = *(const uint4*)p;
  f.q[1] = *(const uint4*)(p + 16);
  return f.v;
}
// B operand (32x16 bf16) from global f32 weights, p -> &W[n*K + kb + 16*half] (16 consecutive f32)
__device__ __forceinline__ v16bf bfrag_g(const float* p) {
  FragU f;
  const float4* p4 = (const float4*)p;
  float4 a = p4[0], b = p4[1], c = p4[2], d = p4[3];
  f.u[0] = pack2(a.x, a.y); f.u[1] = pack2(a.z, a.w);
  f.u[2] = pack2(b.x, b.y); f.u[3] = pack2(b.z, b.w);
  f.u[4] = pack2(c.x, c.y); f.u[5] = pack2(c.z, c.w);
  f.u[6] = pack2(d.x, d.y); f.u[7] = pack2(d.z, d.w);
  return f.v;
}
// stage 32 consecutive f32 -> 32 bf16 (64B) into LDS
__device__ __forceinline__ void stage_row32(char* dst, const float* src) {
#pragma unroll
  for (int j = 0; j < 8; j++) {
    float4 v = ((const float4*)src)[j];
    ((uint2*)dst)[j] = make_uint2(pack2(v.x, v.y), pack2(v.z, v.w));
  }
}
// CDNA5 async global->LDS copy, 16B per lane (ASYNCcnt tracked)
__device__ __forceinline__ void async_g2l_b128(unsigned lds_addr, unsigned long long gaddr) {
  asm volatile("global_load_async_to_lds_b128 %0, %1, off"
               :: "v"(lds_addr), "v"(gaddr) : "memory");
}
__device__ __forceinline__ void wait_async0() {
  asm volatile("s_wait_asynccnt 0" ::: "memory");
}

// ---------------- small kernels ----------------
__global__ void make_masks_k(const int* nm, float* xmf, float* qscale) {
  int i = blockIdx.x * 256 + threadIdx.x;
  float m = nm[i] ? 1.f : 0.f;
  xmf[i] = m;
  qscale[i] = m * 0.17677669529663687f; // mask * 1/sqrt(DF)
}

__global__ void pool_x_k(const float* X, float* xz) {
  int b = blockIdx.x, c = threadIdx.x;
  const float* p = X + (size_t)b * NNODE * 256 + c;
  float s = 0.f, s2 = 0.f, mn = 3e38f, mx = -3e38f;
  for (int i = 0; i < NNODE; i++) {
    float v = p[(size_t)i * 256];
    s += v; s2 += v * v; mn = fminf(mn, v); mx = fmaxf(mx, v);
  }
  float mean = s / (float)NNODE;
  float var = (s2 - s * s / (float)NNODE) / (float)(NNODE - 1);
  float* o = xz + (size_t)b * 1024;
  o[c] = mean; o[256 + c] = mn; o[512 + c] = mx; o[768 + c] = sqrtf(fmaxf(var, 0.f));
}

__global__ void pool_e_part_k(const float* E, float* part) {
  int wg = blockIdx.x, b = blockIdx.y;
  int t = threadIdx.x, ch = t & 63, rs = t >> 6;
  size_t r0 = (size_t)wg * 1024;
  float s = 0.f, s2 = 0.f, mn = 3e38f, mx = -3e38f;
  for (int it = 0; it < 256; it++) {
    size_t row = r0 + rs + 4 * (size_t)it;
    const float* p = &E[((size_t)b * 65536 + row) * 64 + ch];
    __builtin_prefetch(p + 16 * 64, 0, 0);
    float v = *p;
    s += v; s2 += v * v; mn = fminf(mn, v); mx = fmaxf(mx, v);
  }
  __shared__ float sh[256][4];
  sh[t][0] = s; sh[t][1] = mn; sh[t][2] = mx; sh[t][3] = s2;
  __syncthreads();
  if (t < 64) {
    float S = 0.f, S2 = 0.f, MN = 3e38f, MX = -3e38f;
    for (int g = 0; g < 4; g++) {
      int u = g * 64 + t;
      S += sh[u][0]; MN = fminf(MN, sh[u][1]); MX = fmaxf(MX, sh[u][2]); S2 += sh[u][3];
    }
    float* o = part + (((size_t)b * 64 + wg) * 64 + t) * 4;
    o[0] = S; o[1] = MN; o[2] = MX; o[3] = S2;
  }
}

struct YParams {
  const float *y, *xz, *ezpart;
  const float *Wyy, *byy, *Wxy, *bxy, *Wey, *bey;
  const float *W1, *b1, *W2, *b2;
  const float *g1, *be1, *Wl1, *bl1, *Wl2, *bl2, *g2, *be2;
  const float *Wem, *bem, *Wea, *bea, *Wxm, *bxm, *Wxa, *bxa;
  float *yem, *yea, *yxm, *yxa, *out_yo;
};

__global__ void y_path_k(YParams P) {
  int b = blockIdx.x, t = threadIdx.x;
  __shared__ float ez_s[256], sums_s[64], ny[64], t1[64], yr_s[64], ffv[64], h[2048];
  {
    int ch = t & 63, stat = t >> 6;
    float acc = (stat == 1) ? 3e38f : (stat == 2 ? -3e38f : 0.f);
    for (int g = 0; g < 64; g++) {
      float v = P.ezpart[(((size_t)b * 64 + g) * 64 + ch) * 4 + stat];
      if (stat == 1) acc = fminf(acc, v);
      else if (stat == 2) acc = fmaxf(acc, v);
      else acc += v;
    }
    if (stat == 0) sums_s[ch] = acc;
    __syncthreads();
    const float cnt = 65536.f;
    float r;
    if (stat == 0) r = acc / cnt;
    else if (stat == 3) { float S = sums_s[ch]; r = sqrtf(fmaxf((acc - S * S / cnt) / (cnt - 1.f), 0.f)); }
    else r = acc;
    ez_s[stat * 64 + ch] = r;
  }
  __syncthreads();
  const float* yb = P.y + b * 64;
  if (t < 64) {
    float a = P.byy[t] + P.bxy[t] + P.bey[t];
    for (int j = 0; j < 64; j++) a += yb[j] * P.Wyy[t * 64 + j];
    const float* xzb = P.xz + (size_t)b * 1024;
    for (int j = 0; j < 1024; j++) a += xzb[j] * P.Wxy[(size_t)t * 1024 + j];
    for (int j = 0; j < 256; j++) a += ez_s[j] * P.Wey[(size_t)t * 256 + j];
    ny[t] = a;
  }
  __syncthreads();
  if (t < 64) {
    float a = P.b1[t];
    for (int j = 0; j < 64; j++) a += ny[j] * P.W1[t * 64 + j];
    t1[t] = fmaxf(a, 0.f);
  }
  __syncthreads();
  if (t < 64) {
    float a = P.b2[t];
    for (int j = 0; j < 64; j++) a += t1[j] * P.W2[t * 64 + j];
    ny[t] = yb[t] + a;
  }
  __syncthreads();
  {
    float s = 0.f, s2 = 0.f;
    for (int j = 0; j < 64; j++) { float v = ny[j]; s += v; s2 += v * v; }
    float m = s / 64.f, var = s2 / 64.f - m * m, rstd = rsqrtf(var + EPSV);
    if (t < 64) yr_s[t] = (ny[t] - m) * rstd * P.g1[t] + P.be1[t];
  }
  __syncthreads();
#pragma unroll
  for (int j = 0; j < 8; j++) {
    int hj = t * 8 + j;
    float a = P.bl1[hj];
    for (int k = 0; k < 64; k++) a += yr_s[k] * P.Wl1[(size_t)hj * 64 + k];
    h[hj] = fmaxf(a, 0.f);
  }
  __syncthreads();
  if (t < 64) {
    float a = P.bl2[t];
    for (int j = 0; j < 2048; j++) a += h[j] * P.Wl2[(size_t)t * 2048 + j];
    ffv[t] = yr_s[t] + a;
  }
  __syncthreads();
  {
    float s = 0.f, s2 = 0.f;
    for (int j = 0; j < 64; j++) { float v = ffv[j]; s += v; s2 += v * v; }
    float m = s / 64.f, var = s2 / 64.f - m * m, rstd = rsqrtf(var + EPSV);
    if (t < 64) P.out_yo[b * 64 + t] = (ffv[t] - m) * rstd * P.g2[t] + P.be2[t];
  }
  {
    int c = t;
    float em = P.bem[c], ea = P.bea[c], xm = P.bxm[c], xa = P.bxa[c];
    for (int j = 0; j < 64; j++) {
      float yv = yb[j];
      em += yv * P.Wem[c * 64 + j]; ea += yv * P.Wea[c * 64 + j];
      xm += yv * P.Wxm[c * 64 + j]; xa += yv * P.Wxa[c * 64 + j];
    }
    P.yem[b * 256 + c] = em; P.yea[b * 256 + c] = ea;
    P.yxm[b * 256 + c] = xm; P.yxa[b * 256 + c] = xa;
  }
}

// -------------- generic bf16 WMMA GEMM: C = act((A @ W^T + b) * rowscale) --------------
// K must be a power-of-two multiple of 32 here (256 or 2048).
__global__ __launch_bounds__(256) void gemm_k(const float* __restrict__ A,
                                              const float* __restrict__ W,
                                              const float* __restrict__ bias,
                                              const float* __restrict__ rowscale,
                                              float* __restrict__ C,
                                              int M, int N, int K, int flags) {
  extern __shared__ char sm[];
  const int S = 2 * K + 16;
  int tid = threadIdx.x;
  int m0 = blockIdx.x * 16;
  int n0 = blockIdx.y * 128;
  int kq = K >> 2;                       // float4s per row (power of two)
  int ksh = 31 - __clz(kq);
  int nvec = 16 << ksh;
  for (int v = tid; v < nvec; v += 256) {
    int row = v >> ksh, c4 = v & (kq - 1);
    float4 f = ((const float4*)(A + (size_t)(m0 + row) * K))[c4];
    *(uint2*)(sm + row * S + c4 * 8) = make_uint2(pack2(f.x, f.y), pack2(f.z, f.w));
  }
  __syncthreads();
  int w = tid >> 5, lane = tid & 31, hl = lane >> 4, l15 = lane & 15;
  int n = n0 + w * 16 + l15;
  v8f acc = z8();
  for (int kb = 0; kb < K; kb += 32) {
    v16bf af = afrag_s(sm + l15 * S + (kb + 16 * hl) * 2);
    v16bf bf = bfrag_g(W + (size_t)n * K + kb + 16 * hl);
    acc = wmma_bf(af, bf, acc);
  }
  float bn = bias ? bias[n] : 0.f;
  float* Cp = C + (size_t)(m0 + 8 * hl) * N + n;
  const float* rsp = rowscale ? rowscale + m0 + 8 * hl : nullptr;
#pragma unroll
  for (int i = 0; i < 8; i++) {
    float vv = acc[i] + bn;
    if (rsp) vv *= rsp[i];
    if (flags & 1) vv = fmaxf(vv, 0.f);
    Cp[(size_t)i * N] = vv;
  }
  (void)M;
}

// -------------- fused edge mega kernel --------------
struct MegaP {
  const float *E, *Er, *El;
  const float *Wm, *bm, *Wa, *ba, *Wrm, *brm, *Wra, *bra, *Wout, *bout;
  const float *Qf, *Kf, *Vf, *xmf, *yem, *yea;
  float *newE, *part;
};

__global__ __launch_bounds__(512) void edge_mega_k(MegaP P) {
  extern __shared__ char sm[];
  char* Ae = sm;                      // 256*144
  char* Ar = Ae + 256 * 144;
  char* Al = Ar + 256 * 144;
  char* A2 = Al + 256 * 144;          // 256*528 (newE_pre bf16, K=256)
  float* Qs = (float*)(A2 + 256 * 528);
  float* Ks = Qs + 16 * 256;
  float* Vs = Ks + 16 * 256;
  float* xmk = Vs + 16 * 256;
  float* xmqs = xmk + 16;
  float* zfs = xmqs + 16;             // 256

  int qt = blockIdx.x, ks = blockIdx.y, b = blockIdx.z;
  int q0 = qt * 16, k0 = ks * 16;
  int tid = threadIdx.x;
  int w = tid >> 5, lane = tid & 31, hl = lane >> 4, l15 = lane & 15;
  int c = w * 16 + l15;

  // ---- staging (start async Q/K/V copies first so they overlap the cvt work) ----
  {
    unsigned qsb = (unsigned)(unsigned long long)(uintptr_t)Qs;
    unsigned ksb = (unsigned)(unsigned long long)(uintptr_t)Ks;
    unsigned vsb = (unsigned)(unsigned long long)(uintptr_t)Vs;
    unsigned long long qg = (unsigned long long)(uintptr_t)(P.Qf + ((size_t)b * 256 + q0) * 256);
    unsigned long long kg = (unsigned long long)(uintptr_t)(P.Kf + ((size_t)b * 256 + k0) * 256);
    unsigned long long vg = (unsigned long long)(uintptr_t)(P.Vf + ((size_t)b * 256 + k0) * 256);
#pragma unroll
    for (int v = 0; v < 2; v++) {
      unsigned lo = (unsigned)(tid + v * 512) * 16u;   // 1024 x 16B = 16KB per tile
      async_g2l_b128(qsb + lo, qg + lo);
      async_g2l_b128(ksb + lo, kg + lo);
      async_g2l_b128(vsb + lo, vg + lo);
    }
  }
  if (tid < 16) { xmk[tid] = P.xmf[b * 256 + k0 + tid]; xmqs[tid] = P.xmf[b * 256 + q0 + tid]; }

  // resident projection-weight fragments (e_mul, e_add, e_reg_mul, e_reg_add)
  v16bf Wm0 = bfrag_g(P.Wm + (size_t)c * 64 + 16 * hl);
  v16bf Wm1 = bfrag_g(P.Wm + (size_t)c * 64 + 32 + 16 * hl);
  v16bf Wa0 = bfrag_g(P.Wa + (size_t)c * 64 + 16 * hl);
  v16bf Wa1 = bfrag_g(P.Wa + (size_t)c * 64 + 32 + 16 * hl);
  v16bf Wr0 = bfrag_g(P.Wrm + (size_t)c * 64 + 16 * hl);
  v16bf Wr1 = bfrag_g(P.Wrm + (size_t)c * 64 + 32 + 16 * hl);
  v16bf Ws0 = bfrag_g(P.Wra + (size_t)c * 64 + 16 * hl);
  v16bf Ws1 = bfrag_g(P.Wra + (size_t)c * 64 + 32 + 16 * hl);
  float sbm = P.bm[c], sba = P.ba[c], sbrm = P.brm[c], sbra = P.bra[c];
  float yemc = P.yem[b * 256 + c], yeac = P.yea[b * 256 + c];

  // E / E_reg / E_LLM tiles -> bf16 LDS (converted through VGPRs)
  {
    int row = tid >> 1, hf = tid & 1;
    size_t gidx = (((size_t)b * 256 + q0 + (row >> 4)) * 256 + (k0 + (row & 15))) * 64 + hf * 32;
    stage_row32(Ae + row * 144 + hf * 64, P.E + gidx);
    stage_row32(Ar + row * 144 + hf * 64, P.Er + gidx);
    if (hf == 0) zfs[row] = (P.Er[gidx] == 0.f) ? 0.f : 1.f;
    stage_row32(Al + row * 144 + hf * 64, P.El + gidx);
  }
  wait_async0();
  __syncthreads();

  // ---- per q-row (M-tile) FiLM + attention partials + newE_pre ----
#pragma unroll 1
  for (int mt = 0; mt < 16; mt++) {
    const int rbase = (mt * 16 + l15) * 144 + 32 * hl;
    float qv = Qs[mt * 256 + c];   // masked & scaled
    float xmq = xmqs[mt];
    float Yv[8];

    v16bf a0 = afrag_s(Ae + rbase);
    v16bf a1 = afrag_s(Ae + rbase + 64);
    v8f cm = wmma_bf(a1, Wm1, wmma_bf(a0, Wm0, z8()));
    v8f ca = wmma_bf(a1, Wa1, wmma_bf(a0, Wa0, z8()));
#pragma unroll
    for (int i = 0; i < 8; i++) {
      int kk = i + 8 * hl;
      float em = xmq * xmk[kk];
      float y0 = qv * Ks[kk * 256 + c];
      float e1 = (cm[i] + sbm) * em, e2 = (ca[i] + sba) * em;
      Yv[i] = y0 * (e1 + 1.f) + e2;
    }
    a0 = afrag_s(Ar + rbase);
    a1 = afrag_s(Ar + rbase + 64);
    cm = wmma_bf(a1, Wr1, wmma_bf(a0, Wr0, z8()));
    ca = wmma_bf(a1, Ws1, wmma_bf(a0, Ws0, z8()));
#pragma unroll
    for (int i = 0; i < 8; i++) {
      int kk = i + 8 * hl;
      float em = xmq * xmk[kk] * zfs[mt * 16 + kk];
      float r1 = (cm[i] + sbrm) * em, r2 = (ca[i] + sbra) * em;
      Yv[i] = Yv[i] * (r1 + 1.f) + r2;
    }
    a0 = afrag_s(Al + rbase);
    a1 = afrag_s(Al + rbase + 64);
    cm = wmma_bf(a1, Wr1, wmma_bf(a0, Wr0, z8()));   // shared e_reg weights (reference reuse)
    ca = wmma_bf(a1, Ws1, wmma_bf(a0, Ws0, z8()));
#pragma unroll
    for (int i = 0; i < 8; i++) {
      int kk = i + 8 * hl;
      float em = xmq * xmk[kk];
      float l1 = (cm[i] + sbrm) * em, l2 = (ca[i] + sbra) * em;
      Yv[i] = Yv[i] * (l1 + 1.f) + l2;
    }
    // attention partial (split-k softmax)
    float mx = -3e38f;
#pragma unroll
    for (int i = 0; i < 8; i++) {
      int kk = i + 8 * hl;
      if (xmk[kk] > 0.5f) mx = fmaxf(mx, Yv[i]);
    }
    mx = fmaxf(mx, __shfl_xor(mx, 16, 32));
    float sp = 0.f, ap = 0.f;
#pragma unroll
    for (int i = 0; i < 8; i++) {
      int kk = i + 8 * hl;
      float p = (xmk[kk] > 0.5f) ? __expf(Yv[i] - mx) : 0.f;
      sp += p; ap += p * Vs[kk * 256 + c];
    }
    sp += __shfl_xor(sp, 16, 32);
    ap += __shfl_xor(ap, 16, 32);
    if (hl == 0) {
      size_t pb = ((((size_t)b * 256 + q0 + mt) * 256 + c) * 16 + ks) * 3;
      P.part[pb] = mx; P.part[pb + 1] = sp; P.part[pb + 2] = ap;
    }
    // newE_pre = yea + (yem+1)*Y -> LDS bf16 (A operand of e_out GEMM)
#pragma unroll
    for (int i = 0; i < 8; i++) {
      int kk = i + 8 * hl;
      float ne = yeac + (yemc + 1.f) * Yv[i];
      *(unsigned short*)(A2 + (mt * 16 + kk) * 528 + c * 2) = (unsigned short)bfbits(ne);
    }
  }
  __syncthreads();

  // ---- e_out GEMM (K=256 -> 64) + mask + store newE ----
  {
    int o = (w & 3) * 16 + l15;
    float bo = P.bout[o];
    v16bf wb[8];
#pragma unroll
    for (int kc = 0; kc < 8; kc++)
      wb[kc] = bfrag_g(P.Wout + (size_t)o * 256 + kc * 32 + 16 * hl);
#pragma unroll
    for (int s = 0; s < 4; s++) {
      int mt = (w >> 2) * 4 + s;
      v8f acc = z8();
#pragma unroll
      for (int kc = 0; kc < 8; kc++) {
        v16bf af = afrag_s(A2 + (mt * 16 + l15) * 528 + (kc * 32 + 16 * hl) * 2);
        acc = wmma_bf(af, wb[kc], acc);
      }
      float xmq = xmqs[mt];
#pragma unroll
      for (int i = 0; i < 8; i++) {
        int kk = i + 8 * hl;
        float em = xmq * xmk[kk];
        size_t oi = (((size_t)b * 256 + q0 + mt) * 256 + (k0 + kk)) * 64 + o;
        P.newE[oi] = (acc[i] + bo) * em;
      }
    }
  }
}

__global__ void attn_merge_k(const float* __restrict__ part, float* __restrict__ wV) {
  size_t idx = (size_t)blockIdx.x * 256 + threadIdx.x;
  const float* p = part + idx * 48;
  float m = -3e38f;
  for (int i = 0; i < 16; i++) m = fmaxf(m, p[i * 3]);
  float s = 0.f, a = 0.f;
  for (int i = 0; i < 16; i++) {
    float sc = __expf(p[i * 3] - m);
    s += p[i * 3 + 1] * sc; a += p[i * 3 + 2] * sc;
  }
  wV[idx] = (s > 0.f) ? a / s : 0.f;
}

__global__ void nx_prep_k(const float* wV, const float* yxm, const float* yxa, float* nx) {
  size_t idx = (size_t)blockIdx.x * 256 + threadIdx.x;
  int c = idx & 255, b = (int)(idx >> 16);
  nx[idx] = yxa[b * 256 + c] + (yxm[b * 256 + c] + 1.f) * wV[idx];
}

__global__ void ln256_k(const float* A, const float* Badd, const float* g, const float* be, float* out) {
  __shared__ float red[8];
  int r = blockIdx.x, t = threadIdx.x;
  float v = A[(size_t)r * 256 + t] + Badd[(size_t)r * 256 + t];
  float s = v;
  for (int o = 16; o; o >>= 1) s += __shfl_xor(s, o, 32);
  if ((t & 31) == 0) red[t >> 5] = s;
  __syncthreads();
  float tot = 0.f;
  for (int i = 0; i < 8; i++) tot += red[i];
  float m = tot / 256.f, d = v - m;
  __syncthreads();
  float s2 = d * d;
  for (int o = 16; o; o >>= 1) s2 += __shfl_xor(s2, o, 32);
  if ((t & 31) == 0) red[t >> 5] = s2;
  __syncthreads();
  float tv = 0.f;
  for (int i = 0; i < 8; i++) tv += red[i];
  float rstd = rsqrtf(tv / 256.f + EPSV);
  out[(size_t)r * 256 + t] = d * rstd * g[t] + be[t];
}

// -------------- fused E residual-LN + FFE + LN --------------
struct EffnP {
  const float *E, *newE, *g1, *be1, *W1, *b1, *W2, *b2, *g2, *be2;
  float* out;
};

__global__ __launch_bounds__(256) void e_ffn_k(EffnP P) {
  extern __shared__ char sm[];
  char* Ae = sm;                             // 256*144 bf16 normalized rows
  float* ErS = (float*)(Ae + 256 * 144);     // 256*68 f32
  char* Hs = (char*)(ErS + 256 * 68);        // 256*272 bf16 hidden
  float* Out = (float*)(Hs + 256 * 272);     // 256*68 f32
  size_t R0 = (size_t)blockIdx.x * 256;
  int tid = threadIdx.x;
  int l15w = tid & 15, hw = tid >> 4;

  // Phase A: LN1(E + newE), 16 lanes per row
  for (int it = 0; it < 16; it++) {
    int row = it * 16 + hw;
    float4 e4 = ((const float4*)(P.E + (R0 + row) * 64))[l15w];
    float4 n4 = ((const float4*)(P.newE + (R0 + row) * 64))[l15w];
    float4 v;
    v.x = e4.x + n4.x; v.y = e4.y + n4.y; v.z = e4.z + n4.z; v.w = e4.w + n4.w;
    float s = v.x + v.y + v.z + v.w;
    float s2 = v.x * v.x + v.y * v.y + v.z * v.z + v.w * v.w;
    for (int o = 1; o < 16; o <<= 1) { s += __shfl_xor(s, o, 32); s2 += __shfl_xor(s2, o, 32); }
    float m = s / 64.f, rstd = rsqrtf(fmaxf(s2 / 64.f - m * m, 0.f) + EPSV);
    int cb = l15w * 4;
    float4 nr;
    nr.x = (v.x - m) * rstd * P.g1[cb]     + P.be1[cb];
    nr.y = (v.y - m) * rstd * P.g1[cb + 1] + P.be1[cb + 1];
    nr.z = (v.z - m) * rstd * P.g1[cb + 2] + P.be1[cb + 2];
    nr.w = (v.w - m) * rstd * P.g1[cb + 3] + P.be1[cb + 3];
    ((float4*)(ErS + row * 68))[l15w] = nr;
    *(uint2*)(Ae + row * 144 + l15w * 8) = make_uint2(pack2(nr.x, nr.y), pack2(nr.z, nr.w));
  }
  __syncthreads();
  int w = tid >> 5, lane = tid & 31, hl = lane >> 4, l15 = lane & 15;
  // GEMM1: 64 -> 128, relu
  {
    int n = w * 16 + l15;
    v16bf wb0 = bfrag_g(P.W1 + (size_t)n * 64 + 16 * hl);
    v16bf wb1 = bfrag_g(P.W1 + (size_t)n * 64 + 32 + 16 * hl);
    float bn = P.b1[n];
    for (int mt = 0; mt < 16; mt++) {
      const int rb = (mt * 16 + l15) * 144 + 32 * hl;
      v16bf a0 = afrag_s(Ae + rb);
      v16bf a1 = afrag_s(Ae + rb + 64);
      v8f acc = wmma_bf(a1, wb1, wmma_bf(a0, wb0, z8()));
#pragma unroll
      for (int i = 0; i < 8; i++) {
        int row = mt * 16 + i + 8 * hl;
        float hv = fmaxf(acc[i] + bn, 0.f);
        *(unsigned short*)(Hs + row * 272 + n * 2) = (unsigned short)bfbits(hv);
      }
    }
  }
  __syncthreads();
  // GEMM2: 128 -> 64, + bias + Er residual
  {
    int o = (w & 3) * 16 + l15;
    v16bf wb[4];
#pragma unroll
    for (int kc = 0; kc < 4; kc++)
      wb[kc] = bfrag_g(P.W2 + (size_t)o * 128 + kc * 32 + 16 * hl);
    float bo = P.b2[o];
    for (int s = 0; s < 8; s++) {
      int mt = (w >> 2) * 8 + s;
      v8f acc = z8();
#pragma unroll
      for (int kc = 0; kc < 4; kc++) {
        v16bf af = afrag_s(Hs + (mt * 16 + l15) * 272 + (kc * 32 + 16 * hl) * 2);
        acc = wmma_bf(af, wb[kc], acc);
      }
#pragma unroll
      for (int i = 0; i < 8; i++) {
        int row = mt * 16 + i + 8 * hl;
        Out[row * 68 + o] = acc[i] + bo + ErS[row * 68 + o];
      }
    }
  }
  __syncthreads();
  // Phase D: LN2 -> Eo
  for (int it = 0; it < 16; it++) {
    int row = it * 16 + hw;
    float4 v = ((const float4*)(Out + row * 68))[l15w];
    float s = v.x + v.y + v.z + v.w;
    float s2 = v.x * v.x + v.y * v.y + v.z * v.z + v.w * v.w;
    for (int o = 1; o < 16; o <<= 1) { s += __shfl_xor(s, o, 32); s2 += __shfl_xor(s2, o, 32); }
    float m = s / 64.f, rstd = rsqrtf(fmaxf(s2 / 64.f - m * m, 0.f) + EPSV);
    int cb = l15w * 4;
    float4 nr;
    nr.x = (v.x - m) * rstd * P.g2[cb]     + P.be2[cb];
    nr.y = (v.y - m) * rstd * P.g2[cb + 1] + P.be2[cb + 1];
    nr.z = (v.z - m) * rstd * P.g2[cb + 2] + P.be2[cb + 2];
    nr.w = (v.w - m) * rstd * P.g2[cb + 3] + P.be2[cb + 3];
    ((float4*)(P.out + (R0 + row) * 64))[l15w] = nr;
  }
}

// ---------------- host launcher ----------------
extern "C" void kernel_launch(void* const* d_in, const int* in_sizes, int n_in,
                              void* d_out, int out_size, void* d_ws, size_t ws_size,
                              hipStream_t stream) {
  (void)in_sizes; (void)n_in; (void)out_size; (void)ws_size;
  const float* X    = (const float*)d_in[0];
  const float* E    = (const float*)d_in[1];
  const float* yv   = (const float*)d_in[2];
  const int*   nm   = (const int*)d_in[3];
  const float* Ereg = (const float*)d_in[4];
  const float* Ellm = (const float*)d_in[5];
  auto W = [&](int i) { return (const float*)d_in[i]; };

  char* ws = (char*)d_ws;
  size_t off = 0;
  auto alloc = [&](size_t bytes) { size_t o = off; off += (bytes + 255) & ~(size_t)255; return o; };
  float* xz     = (float*)(ws + alloc(2 * 1024 * 4));
  float* ezpart = (float*)(ws + alloc(2 * 64 * 64 * 4 * 4));
  float* yem    = (float*)(ws + alloc(2 * 256 * 4));
  float* yea    = (float*)(ws + alloc(2 * 256 * 4));
  float* yxm    = (float*)(ws + alloc(2 * 256 * 4));
  float* yxa    = (float*)(ws + alloc(2 * 256 * 4));
  float* xmf    = (float*)(ws + alloc(512 * 4));
  float* qscale = (float*)(ws + alloc(512 * 4));
  float* Qf     = (float*)(ws + alloc(512 * 256 * 4));
  float* Kf     = (float*)(ws + alloc(512 * 256 * 4));
  float* Vf     = (float*)(ws + alloc(512 * 256 * 4));
  float* newE   = (float*)(ws + alloc((size_t)2 * 256 * 256 * 64 * 4));
  float* part   = (float*)(ws + alloc((size_t)2 * 256 * 256 * 16 * 3 * 4));
  float* wV     = (float*)(ws + alloc(512 * 256 * 4));
  float* nxpre  = (float*)(ws + alloc(512 * 256 * 4));
  float* newX   = (float*)(ws + alloc(512 * 256 * 4));
  float* Xr     = (float*)(ws + alloc(512 * 256 * 4));
  float* Hbuf   = (float*)(ws + alloc((size_t)512 * 2048 * 4));
  float* X2     = (float*)(ws + alloc(512 * 256 * 4));

  float* out = (float*)d_out;      // [Xo 131072][Eo 8388608][yo 128]
  float* outEo = out + 131072;
  float* outYo = out + 131072 + 8388608;

  make_masks_k<<<2, 256, 0, stream>>>(nm, xmf, qscale);
  pool_x_k<<<2, 256, 0, stream>>>(X, xz);
  pool_e_part_k<<<dim3(64, 2), 256, 0, stream>>>(E, ezpart);

  YParams yp;
  yp.y = yv; yp.xz = xz; yp.ezpart = ezpart;
  yp.Wyy = W(P_Y_Y);     yp.byy = W(P_Y_Y + 1);
  yp.Wxy = W(P_X_Y);     yp.bxy = W(P_X_Y + 1);
  yp.Wey = W(P_E_Y);     yp.bey = W(P_E_Y + 1);
  yp.W1  = W(P_Y_OUT1);  yp.b1  = W(P_Y_OUT1 + 1);
  yp.W2  = W(P_Y_OUT2);  yp.b2  = W(P_Y_OUT2 + 1);
  yp.g1  = W(P_NORMY1);  yp.be1 = W(P_NORMY1 + 1);
  yp.Wl1 = W(P_LINY1);   yp.bl1 = W(P_LINY1 + 1);
  yp.Wl2 = W(P_LINY2);   yp.bl2 = W(P_LINY2 + 1);
  yp.g2  = W(P_NORMY2);  yp.be2 = W(P_NORMY2 + 1);
  yp.Wem = W(P_Y_E_MUL); yp.bem = W(P_Y_E_MUL + 1);
  yp.Wea = W(P_Y_E_ADD); yp.bea = W(P_Y_E_ADD + 1);
  yp.Wxm = W(P_Y_X_MUL); yp.bxm = W(P_Y_X_MUL + 1);
  yp.Wxa = W(P_Y_X_ADD); yp.bxa = W(P_Y_X_ADD + 1);
  yp.yem = yem; yp.yea = yea; yp.yxm = yxm; yp.yxa = yxa; yp.out_yo = outYo;
  y_path_k<<<2, 256, 0, stream>>>(yp);

  // Q, K, V projections (mask & 1/sqrt(DF) folded via rowscale)
  size_t smA256 = 16 * (2 * 256 + 16);
  gemm_k<<<dim3(32, 2), 256, smA256, stream>>>(X, W(P_Q), W(P_Q + 1), qscale, Qf, 512, 256, 256, 0);
  gemm_k<<<dim3(32, 2), 256, smA256, stream>>>(X, W(P_K), W(P_K + 1), xmf,    Kf, 512, 256, 256, 0);
  gemm_k<<<dim3(32, 2), 256, smA256, stream>>>(X, W(P_V), W(P_V + 1), xmf,    Vf, 512, 256, 256, 0);

  MegaP mp;
  mp.E = E; mp.Er = Ereg; mp.El = Ellm;
  mp.Wm  = W(P_E_MUL);     mp.bm  = W(P_E_MUL + 1);
  mp.Wa  = W(P_E_ADD);     mp.ba  = W(P_E_ADD + 1);
  mp.Wrm = W(P_E_REG_MUL); mp.brm = W(P_E_REG_MUL + 1);
  mp.Wra = W(P_E_REG_ADD); mp.bra = W(P_E_REG_ADD + 1);
  mp.Wout = W(P_E_OUT);    mp.bout = W(P_E_OUT + 1);
  mp.Qf = Qf; mp.Kf = Kf; mp.Vf = Vf; mp.xmf = xmf; mp.yem = yem; mp.yea = yea;
  mp.newE = newE; mp.part = part;
  size_t megaSm = 3 * 256 * 144 + 256 * 528 + 3 * 16 * 256 * 4 + (16 + 16 + 256) * 4;
  edge_mega_k<<<dim3(16, 16, 2), 512, megaSm, stream>>>(mp);

  attn_merge_k<<<512, 256, 0, stream>>>(part, wV);
  nx_prep_k<<<512, 256, 0, stream>>>(wV, yxm, yxa, nxpre);

  gemm_k<<<dim3(32, 2), 256, smA256, stream>>>(nxpre, W(P_X_OUT), W(P_X_OUT + 1), xmf, newX, 512, 256, 256, 0);
  ln256_k<<<512, 256, 0, stream>>>(X, newX, W(P_NORMX1), W(P_NORMX1 + 1), Xr);
  gemm_k<<<dim3(32, 16), 256, smA256, stream>>>(Xr, W(P_LINX1), W(P_LINX1 + 1), nullptr, Hbuf, 512, 2048, 256, 1);
  size_t smA2048 = 16 * (2 * 2048 + 16);
  gemm_k<<<dim3(32, 2), 256, smA2048, stream>>>(Hbuf, W(P_LINX2), W(P_LINX2 + 1), nullptr, X2, 512, 256, 2048, 0);
  ln256_k<<<512, 256, 0, stream>>>(Xr, X2, W(P_NORMX2), W(P_NORMX2 + 1), out);

  EffnP ep;
  ep.E = E; ep.newE = newE;
  ep.g1 = W(P_NORME1); ep.be1 = W(P_NORME1 + 1);
  ep.W1 = W(P_LINE1);  ep.b1  = W(P_LINE1 + 1);
  ep.W2 = W(P_LINE2);  ep.b2  = W(P_LINE2 + 1);
  ep.g2 = W(P_NORME2); ep.be2 = W(P_NORME2 + 1);
  ep.out = outEo;
  size_t effnSm = 256 * 144 + 256 * 68 * 4 + 256 * 272 + 256 * 68 * 4;
  e_ffn_k<<<512, 256, effnSm, stream>>>(ep);
}
